// MultiHeadAttention_85177791414644
// MI455X (gfx1250) — matrix operand order repaired and checked by reading, not compiled, not run on hardware
//
#include <hip/hip_runtime.h>

#define EMB   1024
#define NHEADS 16
#define HDIM   64
#define BATCH   2
#define SEQT 2048
#define SEQS 2048
#define NEG_BIG  (-3.0e38f)
#define MASK_NEG (-3.4028234e38f)

typedef __attribute__((ext_vector_type(16))) _Float16 v16h;
typedef __attribute__((ext_vector_type(8)))  float    v8f;

#define WMMA_F16(a, b, c) \
  __builtin_amdgcn_wmma_f32_16x16x32_f16(false, (a), false, (b), (short)0, (c), false, false)

// ---- fragment loaders (contiguous K per lane half, same k order for A and B) ----

// A 16x32 f16: lanes 0-15 row M=lane hold K 0..15; lanes 16-31 hold K 16..31
static __device__ __forceinline__ v16h load_a16x32(const _Float16* __restrict__ A, int lda,
                                                   int rowbase, int kbase, int lane) {
  const int r   = rowbase + (lane & 15);
  const int klo = kbase + ((lane >> 4) << 4);
  const _Float16* p = A + (size_t)r * lda + klo;
  v16h a;
#pragma unroll
  for (int j = 0; j < 16; ++j) a[j] = p[j];
  return a;
}

// B 32x16: lane holds column n; lanes 0-15 K=kbase..+15, lanes 16-31 K=kbase+16..+31.
// Source is row-major [N][ld] (i.e. B[k][n] = W[n][k]) -> 16 contiguous K per lane.
static __device__ __forceinline__ v16h load_b_colK(const _Float16* __restrict__ W, int ld,
                                                   int nbase, int kbase, int lane) {
  const int n  = nbase + (lane & 15);
  const int k0 = kbase + ((lane >> 4) << 4);
  const _Float16* p = W + (size_t)n * ld + k0;
  v16h b;
#pragma unroll
  for (int j = 0; j < 16; ++j) b[j] = p[j];
  return b;
}

// B 32x16 where source is row-major [K][ld] (B[k][n] = M[k][n]) -> strided K per lane.
static __device__ __forceinline__ v16h load_b_rowK(const _Float16* __restrict__ M, int ld,
                                                   int nbase, int lane) {
  const int n  = nbase + (lane & 15);
  const int k0 = (lane >> 4) << 4;
  v16h b;
#pragma unroll
  for (int j = 0; j < 16; ++j) b[j] = M[(size_t)(k0 + j) * ld + n];
  return b;
}

// ---- kernels ----

__global__ void cvt_f32_to_f16(const float* __restrict__ src, _Float16* __restrict__ dst, int n) {
  for (int i = blockIdx.x * blockDim.x + threadIdx.x; i < n; i += blockDim.x * gridDim.x)
    dst[i] = (_Float16)src[i];
}

// qkv = X @ W^T + b. Register-blocked 64(M)x32(N) per wave: 4 A-frags x 2 B-frags ->
// 8 WMMAs per K-step from 6 fragment loads (~21 FLOP/byte of L2 traffic), pressure
// ~180 VGPRs incl. pipelined loads -> no spills. SCALING folded into Q.
__global__ void __launch_bounds__(256)
qkv_proj_wmma(const _Float16* __restrict__ xq, const _Float16* __restrict__ xk,
              const _Float16* __restrict__ xv, const _Float16* __restrict__ wqkv,
              const float* __restrict__ bqkv,
              _Float16* __restrict__ qp, _Float16* __restrict__ kp,
              _Float16* __restrict__ vp) {
  const int lane = threadIdx.x & 31;
  const int wave = blockIdx.x * (blockDim.x >> 5) + (threadIdx.x >> 5);
  const int TM = (BATCH * SEQT) / 64;  // 64
  const int TN = EMB / 32;             // 32
  const int mat = wave / (TM * TN);
  if (mat >= 3) return;
  const int rem = wave - mat * TM * TN;
  const int tm = rem / TN, tn = rem % TN;
  const _Float16* X = (mat == 0) ? xq : (mat == 1) ? xk : xv;
  const _Float16* W = wqkv + (size_t)mat * EMB * EMB;

  v8f acc[4][2];
#pragma unroll
  for (int mi = 0; mi < 4; ++mi)
#pragma unroll
    for (int ni = 0; ni < 2; ++ni) acc[mi][ni] = (v8f){};

#pragma unroll 1
  for (int k = 0; k < EMB; k += 32) {
    v16h af[4], bf[2];
#pragma unroll
    for (int mi = 0; mi < 4; ++mi) af[mi] = load_a16x32(X, EMB, tm * 64 + mi * 16, k, lane);
#pragma unroll
    for (int ni = 0; ni < 2; ++ni) bf[ni] = load_b_colK(W, EMB, tn * 32 + ni * 16, k, lane);
#pragma unroll
    for (int mi = 0; mi < 4; ++mi)
#pragma unroll
      for (int ni = 0; ni < 2; ++ni) acc[mi][ni] = WMMA_F16(af[mi], bf[ni], acc[mi][ni]);
  }

  const int hi = lane >> 4, ln = lane & 15;
  const float scl = (mat == 0) ? 0.125f : 1.0f;  // HEAD_DIM^-0.5 folded into Q
  _Float16* O = (mat == 0) ? qp : (mat == 1) ? kp : vp;
#pragma unroll
  for (int ni = 0; ni < 2; ++ni) {
    const int n = tn * 32 + ni * 16 + ln;     // 32-wide N tile stays inside one head
    const int h = n >> 6, d = n & (HDIM - 1);
    const float bias = bqkv[mat * EMB + n];
#pragma unroll
    for (int mi = 0; mi < 4; ++mi) {
#pragma unroll
      for (int vi = 0; vi < 8; ++vi) {
        const int m  = tm * 64 + mi * 16 + vi + (hi << 3);
        const int bb = m >> 11, t = m & (SEQT - 1);
        O[((((size_t)bb * NHEADS + h) * SEQS) + t) * HDIM + d] =
            (_Float16)((acc[mi][ni][vi] + bias) * scl);
      }
    }
  }
}

// Flash attention: one wave per (b,h,16-row tile). Online softmax, P staged via LDS for
// the D-layout -> A-layout transpose, P@V accumulated with WMMA.
__global__ void flash_attn_wmma(const _Float16* __restrict__ qp, const _Float16* __restrict__ kp,
                                const _Float16* __restrict__ vp, const float* __restrict__ bias2d,
                                const unsigned char* __restrict__ kmask,
                                _Float16* __restrict__ ctx) {
  __shared__ _Float16 pbuf[8][16 * 32];
  const int lane  = threadIdx.x & 31;
  const int wslot = threadIdx.x >> 5;
  const int wave  = blockIdx.x * (blockDim.x >> 5) + wslot;
  const int tt = wave & (SEQT / 16 - 1);  // 0..127
  const int bh = wave >> 7;
  const int b = bh >> 4, h = bh & 15;
  const int hi = lane >> 4, ln = lane & 15;
  const _Float16* Q = qp + (((size_t)b * NHEADS + h) * SEQT) * HDIM;
  const _Float16* K = kp + (((size_t)b * NHEADS + h) * SEQS) * HDIM;
  const _Float16* V = vp + (((size_t)b * NHEADS + h) * SEQS) * HDIM;

  const v16h qa0 = load_a16x32(Q, HDIM, tt * 16, 0, lane);
  const v16h qa1 = load_a16x32(Q, HDIM, tt * 16, 32, lane);

  float mrun[8], lrun[8];
  v8f o0 = {}, o1 = {}, o2 = {}, o3 = {};
#pragma unroll
  for (int vi = 0; vi < 8; ++vi) { mrun[vi] = NEG_BIG; lrun[vi] = 0.0f; }

  for (int s0 = 0; s0 < SEQS; s0 += 32) {
    // scores: Q(16x64) x K^T(64x32) -> two 16x16 f32 tiles
    v16h kb00 = load_b_colK(K, HDIM, s0,      0,  lane);
    v16h kb01 = load_b_colK(K, HDIM, s0,      32, lane);
    v16h kb10 = load_b_colK(K, HDIM, s0 + 16, 0,  lane);
    v16h kb11 = load_b_colK(K, HDIM, s0 + 16, 32, lane);
    v8f sa = {}, sb = {};
    sa = WMMA_F16(qa0, kb00, sa); sa = WMMA_F16(qa1, kb01, sa);
    sb = WMMA_F16(qa0, kb10, sb); sb = WMMA_F16(qa1, kb11, sb);

    const int c0 = s0 + ln, c1 = c0 + 16;
    const bool mk0 = kmask[b * SEQS + c0] != 0;
    const bool mk1 = kmask[b * SEQS + c1] != 0;
#pragma unroll
    for (int vi = 0; vi < 8; ++vi) {
      const int tr = tt * 16 + vi + (hi << 3);
      float x0 = sa[vi] + bias2d[(size_t)tr * SEQS + c0];
      float x1 = sb[vi] + bias2d[(size_t)tr * SEQS + c1];
      if (mk0) x0 = MASK_NEG;
      if (mk1) x1 = MASK_NEG;
      // row max across the 16 lanes holding this row's columns
      float mx = fmaxf(x0, x1);
      mx = fmaxf(mx, __shfl_xor(mx, 1));
      mx = fmaxf(mx, __shfl_xor(mx, 2));
      mx = fmaxf(mx, __shfl_xor(mx, 4));
      mx = fmaxf(mx, __shfl_xor(mx, 8));
      const float mnew  = fmaxf(mrun[vi], mx);
      const float alpha = __expf(mrun[vi] - mnew);
      const float e0 = __expf(x0 - mnew);
      const float e1 = __expf(x1 - mnew);
      float rs = e0 + e1;
      rs += __shfl_xor(rs, 1);
      rs += __shfl_xor(rs, 2);
      rs += __shfl_xor(rs, 4);
      rs += __shfl_xor(rs, 8);
      lrun[vi] = lrun[vi] * alpha + rs;
      mrun[vi] = mnew;
      o0[vi] *= alpha; o1[vi] *= alpha; o2[vi] *= alpha; o3[vi] *= alpha;
      // stage P (16x32) to LDS in plain row-major for the A-fragment reload
      const int r = vi + (hi << 3);
      pbuf[wslot][r * 32 + ln]      = (_Float16)e0;
      pbuf[wslot][r * 32 + 16 + ln] = (_Float16)e1;
    }
    __syncthreads();  // uniform across block (all waves run identical trip count)

    v16h pa;
    {
      const _Float16* pp = &pbuf[wslot][ln * 32 + (hi << 4)];
#pragma unroll
      for (int j = 0; j < 16; ++j) pa[j] = pp[j];
    }
    const _Float16* Vb = V + (size_t)s0 * HDIM;
    v16h vb0 = load_b_rowK(Vb, HDIM, 0,  lane);
    v16h vb1 = load_b_rowK(Vb, HDIM, 16, lane);
    v16h vb2 = load_b_rowK(Vb, HDIM, 32, lane);
    v16h vb3 = load_b_rowK(Vb, HDIM, 48, lane);
    o0 = WMMA_F16(pa, vb0, o0);
    o1 = WMMA_F16(pa, vb1, o1);
    o2 = WMMA_F16(pa, vb2, o2);
    o3 = WMMA_F16(pa, vb3, o3);
    __syncthreads();
  }

#pragma unroll
  for (int vi = 0; vi < 8; ++vi) {
    const float inv = 1.0f / lrun[vi];
    const int t = tt * 16 + vi + (hi << 3);
    _Float16* dst = ctx + ((size_t)b * SEQT + t) * EMB + h * HDIM;
    dst[0 * 16 + ln] = (_Float16)(o0[vi] * inv);
    dst[1 * 16 + ln] = (_Float16)(o1[vi] * inv);
    dst[2 * 16 + ln] = (_Float16)(o2[vi] * inv);
    dst[3 * 16 + ln] = (_Float16)(o3[vi] * inv);
  }
}

// out = ctx @ Wout^T + b ; f32 output. Same 64x32 register blocking.
__global__ void __launch_bounds__(256)
out_proj_wmma(const _Float16* __restrict__ ctx, const _Float16* __restrict__ wo,
              const float* __restrict__ bo, float* __restrict__ out) {
  const int lane = threadIdx.x & 31;
  const int wave = blockIdx.x * (blockDim.x >> 5) + (threadIdx.x >> 5);
  const int TM = (BATCH * SEQT) / 64;  // 64
  const int TN = EMB / 32;             // 32
  const int tm = wave / TN, tn = wave % TN;
  if (tm >= TM) return;

  v8f acc[4][2];
#pragma unroll
  for (int mi = 0; mi < 4; ++mi)
#pragma unroll
    for (int ni = 0; ni < 2; ++ni) acc[mi][ni] = (v8f){};

#pragma unroll 1
  for (int k = 0; k < EMB; k += 32) {
    v16h af[4], bf[2];
#pragma unroll
    for (int mi = 0; mi < 4; ++mi) af[mi] = load_a16x32(ctx, EMB, tm * 64 + mi * 16, k, lane);
#pragma unroll
    for (int ni = 0; ni < 2; ++ni) bf[ni] = load_b_colK(wo, EMB, tn * 32 + ni * 16, k, lane);
#pragma unroll
    for (int mi = 0; mi < 4; ++mi)
#pragma unroll
      for (int ni = 0; ni < 2; ++ni) acc[mi][ni] = WMMA_F16(af[mi], bf[ni], acc[mi][ni]);
  }

  const int hi = lane >> 4, ln = lane & 15;
#pragma unroll
  for (int ni = 0; ni < 2; ++ni) {
    const int n = tn * 32 + ni * 16 + ln;
    const float bias = bo[n];
#pragma unroll
    for (int mi = 0; mi < 4; ++mi) {
#pragma unroll
      for (int vi = 0; vi < 8; ++vi) {
        const int m = tm * 64 + mi * 16 + vi + (hi << 3);
        out[(size_t)m * EMB + n] = acc[mi][ni][vi] + bias;
      }
    }
  }
}

extern "C" void kernel_launch(void* const* d_in, const int* in_sizes, int n_in,
                              void* d_out, int out_size, void* d_ws, size_t ws_size,
                              hipStream_t stream) {
  const float* query     = (const float*)d_in[0];
  const float* key       = (const float*)d_in[1];
  const float* value     = (const float*)d_in[2];
  const float* attn_bias = (const float*)d_in[3];
  const unsigned char* kmask = (const unsigned char*)d_in[4];
  const float* in_w  = (const float*)d_in[5];
  const float* in_b  = (const float*)d_in[6];
  const float* out_w = (const float*)d_in[7];
  const float* out_b = (const float*)d_in[8];
  float* out = (float*)d_out;

  const size_t ACT = (size_t)BATCH * SEQT * EMB;  // 4,194,304
  const size_t W3  = (size_t)3 * EMB * EMB;
  const size_t W1  = (size_t)EMB * EMB;

  _Float16* ws   = (_Float16*)d_ws;
  _Float16* xq   = ws;            // f16 activations
  _Float16* xk   = xq + ACT;
  _Float16* xv   = xk + ACT;
  _Float16* wqkv = xv + ACT;      // f16 weights
  _Float16* wo   = wqkv + W3;
  _Float16* qp   = wo + W1;       // projected q/k/v, [B][H][S][D]
  _Float16* kp   = qp + ACT;
  _Float16* vp   = kp + ACT;
  _Float16* ctx  = vp + ACT;      // attention output, [B][T][E]

  cvt_f32_to_f16<<<2048, 256, 0, stream>>>(query, xq, (int)ACT);
  cvt_f32_to_f16<<<2048, 256, 0, stream>>>(key,   xk, (int)ACT);
  cvt_f32_to_f16<<<2048, 256, 0, stream>>>(value, xv, (int)ACT);
  cvt_f32_to_f16<<<2048, 256, 0, stream>>>(in_w,  wqkv, (int)W3);
  cvt_f32_to_f16<<<1024, 256, 0, stream>>>(out_w, wo,   (int)W1);

  // 3 * 64 * 32 = 6144 waves / 8 per block
  qkv_proj_wmma<<<768, 256, 0, stream>>>(xq, xk, xv, wqkv, in_b, qp, kp, vp);
  // 2 * 16 * 128 = 4096 waves / 8 per block
  flash_attn_wmma<<<512, 256, 0, stream>>>(qp, kp, vp, attn_bias, kmask, ctx);
  // 64 * 32 = 2048 waves / 8 per block
  out_proj_wmma<<<256, 256, 0, stream>>>(ctx, wo, out_b, out);
}
